// Expert_2499670966775
// MI455X (gfx1250) — matrix operand order repaired and checked by reading, not compiled, not run on hardware
//
#include <hip/hip_runtime.h>

// ---------------------------------------------------------------------------
// Types for CDNA5 WMMA (wave32)
// ---------------------------------------------------------------------------
typedef __attribute__((ext_vector_type(16))) __bf16 bf16x16;
typedef __attribute__((ext_vector_type(8)))  float  floatx8;

#define DIM   2048
#define BATCH 4096
#define NEXP  8

// GEMM tiling
#define BM 64
#define BN 128
#define BK 64
#define APAD 8
#define ASTRIDE (BK + APAD)               // 72 bf16 elems per LDS row (144 B)
#define ABUF_ELEMS (BM * ASTRIDE)         // 4608
#define BBUF_ELEMS (BN * ASTRIDE)         // 9216
#define BUF_ELEMS  (ABUF_ELEMS + BBUF_ELEMS)

// float -> bf16, round-to-nearest-even
__device__ __forceinline__ unsigned short f2bf(float f) {
    unsigned u = __float_as_uint(f);
    u += 0x7FFFu + ((u >> 16) & 1u);
    return (unsigned short)(u >> 16);
}

struct U32x8 { uint4 lo, hi; };

// Build a 32-byte (16 x bf16) WMMA fragment from two 16-byte chunks
__device__ __forceinline__ bf16x16 ldfrag(const __bf16* p0, const __bf16* p1) {
    U32x8 t;
    t.lo = *reinterpret_cast<const uint4*>(p0);
    t.hi = *reinterpret_cast<const uint4*>(p1);
    return __builtin_bit_cast(bf16x16, t);
}

// Async global -> LDS copy of 16 bytes per lane (GLOBAL_LOAD_ASYNC_TO_LDS_B128,
// tracked by ASYNCcnt). LDS address = low 32 bits of the generic shared pointer.
__device__ __forceinline__ void async_copy16(const void* gptr, unsigned lds_addr) {
    asm volatile("global_load_async_to_lds_b128 %0, %1, off"
                 :: "v"(lds_addr), "v"(gptr) : "memory");
}

__device__ __forceinline__ void wait_async_and_barrier() {
    asm volatile("s_wait_asynccnt 0x0" ::: "memory");
    __syncthreads();
}

// Stage one BMxBK A tile and BNxBK B tile into LDS (async, 6 ops / thread)
__device__ __forceinline__ void async_tile_copy(const __bf16* __restrict__ X,
                                                const __bf16* __restrict__ Wb,
                                                unsigned short* sA,
                                                unsigned short* sB,
                                                int bm0, int bn0, int k0, int t) {
#pragma unroll
    for (int i = 0; i < 2; ++i) {           // A: 512 chunks of 16B
        const int c   = t + 256 * i;
        const int row = c >> 3;
        const int off = (c & 7) * 8;
        const __bf16* g = X + (size_t)(bm0 + row) * DIM + k0 + off;
        const unsigned lds = (unsigned)(unsigned long long)(sA + row * ASTRIDE + off);
        async_copy16(g, lds);
    }
#pragma unroll
    for (int i = 0; i < 4; ++i) {           // B: 1024 chunks of 16B
        const int c   = t + 256 * i;
        const int row = c >> 3;
        const int off = (c & 7) * 8;
        const __bf16* g = Wb + (size_t)(bn0 + row) * DIM + k0 + off;
        const unsigned lds = (unsigned)(unsigned long long)(sB + row * ASTRIDE + off);
        async_copy16(g, lds);
    }
}

// ---------------------------------------------------------------------------
// Cast input activations f32 -> bf16 (4 elements / thread)
// ---------------------------------------------------------------------------
__global__ void cast_x_kernel(const float* __restrict__ x,
                              unsigned short* __restrict__ xb) {
    size_t i4 = ((size_t)blockIdx.x * blockDim.x + threadIdx.x) * 4;
    const float4 v = *reinterpret_cast<const float4*>(x + i4);
    uint2 o;
    o.x = (unsigned)f2bf(v.x) | ((unsigned)f2bf(v.y) << 16);
    o.y = (unsigned)f2bf(v.z) | ((unsigned)f2bf(v.w) << 16);
    *reinterpret_cast<uint2*>(xb + i4) = o;
}

// ---------------------------------------------------------------------------
// Blend expert weights for one layer: wb[o,k] = bf16( sum_e c[e]*W[l,e,o,k] )
// ---------------------------------------------------------------------------
__global__ void blend_w_kernel(const float* __restrict__ W,
                               const float* __restrict__ c,
                               unsigned short* __restrict__ wb,
                               int layer) {
    const size_t DD = (size_t)DIM * DIM;
    size_t i4 = ((size_t)blockIdx.x * blockDim.x + threadIdx.x) * 4;
    const float* base = W + (size_t)layer * NEXP * DD + i4;

    float4 acc = make_float4(0.f, 0.f, 0.f, 0.f);
#pragma unroll
    for (int e = 0; e < NEXP; ++e) {
        const float4 v = *reinterpret_cast<const float4*>(base + (size_t)e * DD);
        const float ce = c[e];
        acc.x += ce * v.x; acc.y += ce * v.y;
        acc.z += ce * v.z; acc.w += ce * v.w;
    }
    uint2 o;
    o.x = (unsigned)f2bf(acc.x) | ((unsigned)f2bf(acc.y) << 16);
    o.y = (unsigned)f2bf(acc.z) | ((unsigned)f2bf(acc.w) << 16);
    *reinterpret_cast<uint2*>(wb + i4) = o;
}

// ---------------------------------------------------------------------------
// bf16 WMMA GEMM with async-LDS double buffering:
//   out[b,o] = act( sum_k X[b,k]*Wb[o,k] + bias[o] )
// Block = 256 threads = 8 waves (2M x 4N), block tile 64x128, BK=64.
// ---------------------------------------------------------------------------
template <bool DO_ELU, bool OUT_BF16>
__global__ __launch_bounds__(256)
void gemm_wmma_kernel(const __bf16* __restrict__ X,
                      const __bf16* __restrict__ Wb,
                      const float*  __restrict__ bias,
                      unsigned short* __restrict__ out_bf,
                      float* __restrict__ out_f32) {
    __shared__ __align__(16) unsigned short smem[2 * BUF_ELEMS];  // ~54 KB

    const int t    = threadIdx.x;
    const int lane = t & 31;
    const int wave = t >> 5;
    const int wm   = wave & 1;        // 2 waves along M
    const int wn   = wave >> 1;       // 4 waves along N

    const int bm0 = blockIdx.x * BM;
    const int bn0 = blockIdx.y * BN;

    floatx8 acc00 = {0,0,0,0,0,0,0,0};
    floatx8 acc01 = {0,0,0,0,0,0,0,0};
    floatx8 acc10 = {0,0,0,0,0,0,0,0};
    floatx8 acc11 = {0,0,0,0,0,0,0,0};

    // Prologue: stage K block 0 into buffer 0
    async_tile_copy(X, Wb, smem, smem + ABUF_ELEMS, bm0, bn0, 0, t);
    wait_async_and_barrier();

    // Per-lane fragment coordinates inside the LDS tiles.
    // A (16-bit 16x32): lane<16 -> K 0-7 & 16-23, lane>=16 -> K 8-15 & 24-31.
    const int arow  = wm * 32 + (lane & 15);
    const int koffA = (lane >> 4) << 3;   // 0 or 8
    // B (16-bit 32x16): lane<16 -> K 0-7 & 16-23, lane>=16 -> K 8-15 & 24-31 of col N=lane&15.
    const int bcol  = wn * 32 + (lane & 15);
    const int koffB = (lane >> 4) << 3;   // 0 or 8

    const int KT = DIM / BK;   // 32
    for (int kt = 0; kt < KT; ++kt) {
        const int cur = kt & 1;
        unsigned short* sA = smem + cur * BUF_ELEMS;
        unsigned short* sB = sA + ABUF_ELEMS;

        if (kt + 1 < KT) {   // overlap next tile's async copy with compute
            unsigned short* nA = smem + (cur ^ 1) * BUF_ELEMS;
            async_tile_copy(X, Wb, nA, nA + ABUF_ELEMS, bm0, bn0, (kt + 1) * BK, t);
        }

#pragma unroll
        for (int kk = 0; kk < BK; kk += 32) {
            const __bf16* pa0 = (const __bf16*)(sA + arow * ASTRIDE + kk + koffA);
            const __bf16* pa1 = pa0 + 16 * ASTRIDE;
            const __bf16* pb0 = (const __bf16*)(sB + bcol * ASTRIDE + kk + koffB);
            const __bf16* pb1 = pb0 + 16 * ASTRIDE;

            const bf16x16 a0 = ldfrag(pa0, pa0 + 16);
            const bf16x16 a1 = ldfrag(pa1, pa1 + 16);
            const bf16x16 b0 = ldfrag(pb0, pb0 + 16);
            const bf16x16 b1 = ldfrag(pb1, pb1 + 16);

            acc00 = __builtin_amdgcn_wmma_f32_16x16x32_bf16(false, a0, false, b0,
                                                            (short)0, acc00, false, false);
            acc01 = __builtin_amdgcn_wmma_f32_16x16x32_bf16(false, a0, false, b1,
                                                            (short)0, acc01, false, false);
            acc10 = __builtin_amdgcn_wmma_f32_16x16x32_bf16(false, a1, false, b0,
                                                            (short)0, acc10, false, false);
            acc11 = __builtin_amdgcn_wmma_f32_16x16x32_bf16(false, a1, false, b1,
                                                            (short)0, acc11, false, false);
        }

        // Own async writes done -> barrier makes the next buffer visible to all
        // waves and guarantees no one still reads the buffer we overwrite next.
        wait_async_and_barrier();
    }

    // Epilogue. C/D layout: VGPR r -> M = r + 8*(lane>>4), N = lane&15.
    const int m_wave = bm0 + wm * 32;
    const int n_wave = bn0 + wn * 32;
    const int col0   = n_wave + (lane & 15);
    const int col1   = col0 + 16;
    const float bs0  = bias[col0];
    const float bs1  = bias[col1];
    const int rbase  = m_wave + ((lane >> 4) << 3);

    floatx8 accs[2][2] = {{acc00, acc01}, {acc10, acc11}};
#pragma unroll
    for (int mi = 0; mi < 2; ++mi) {
#pragma unroll
        for (int ni = 0; ni < 2; ++ni) {
            const int   col = ni ? col1 : col0;
            const float bs  = ni ? bs1  : bs0;
#pragma unroll
            for (int r = 0; r < 8; ++r) {
                float v = accs[mi][ni][r] + bs;
                if (DO_ELU) v = (v > 0.f) ? v : (__expf(v) - 1.f);
                const size_t idx = (size_t)(rbase + mi * 16 + r) * DIM + col;
                if (OUT_BF16) out_bf[idx]  = f2bf(v);
                else          out_f32[idx] = v;
            }
        }
    }
}

// ---------------------------------------------------------------------------
// Host launch
// ---------------------------------------------------------------------------
extern "C" void kernel_launch(void* const* d_in, const int* in_sizes, int n_in,
                              void* d_out, int out_size, void* d_ws, size_t ws_size,
                              hipStream_t stream) {
    const float* c  = (const float*)d_in[0];               // (8,)
    const float* x  = (const float*)d_in[1];               // (4096, 2048)
    const float* W  = (const float*)d_in[2];               // (3, 8, 2048, 2048)
    const float* Bv = (const float*)d_in[3];               // (3, 2048)
    float* out = (float*)d_out;                            // (4096, 2048) f32

    char* ws = (char*)d_ws;
    unsigned short* wbuf = (unsigned short*)ws;                              //  8 MB
    unsigned short* xa   = (unsigned short*)(ws + (size_t)8  * 1024 * 1024); // 16 MB
    unsigned short* xbuf = (unsigned short*)(ws + (size_t)24 * 1024 * 1024); // 16 MB

    const size_t DD = (size_t)DIM * DIM;        // 4M
    const size_t BD = (size_t)BATCH * DIM;      // 8M

    // x -> bf16
    cast_x_kernel<<<(unsigned)(BD / 1024), 256, 0, stream>>>(x, xa);

    dim3 ggrid(BATCH / BM, DIM / BN, 1);
    dim3 gblk(256, 1, 1);
    unsigned bgrid = (unsigned)(DD / 1024);

    // Layer 0: xa -> xbuf (ELU, bf16 out)
    blend_w_kernel<<<bgrid, 256, 0, stream>>>(W, c, wbuf, 0);
    gemm_wmma_kernel<true, true><<<ggrid, gblk, 0, stream>>>(
        (const __bf16*)xa, (const __bf16*)wbuf, Bv + 0 * DIM, xbuf, nullptr);

    // Layer 1: xbuf -> xa (ELU, bf16 out)
    blend_w_kernel<<<bgrid, 256, 0, stream>>>(W, c, wbuf, 1);
    gemm_wmma_kernel<true, true><<<ggrid, gblk, 0, stream>>>(
        (const __bf16*)xbuf, (const __bf16*)wbuf, Bv + 1 * DIM, xa, nullptr);

    // Layer 2: xa -> out (no act, f32 out)
    blend_w_kernel<<<bgrid, 256, 0, stream>>>(W, c, wbuf, 2);
    gemm_wmma_kernel<false, false><<<ggrid, gblk, 0, stream>>>(
        (const __bf16*)xa, (const __bf16*)wbuf, Bv + 2 * DIM, nullptr, out);
}
